// GatedAttentionLayer_19207093747766
// MI455X (gfx1250) — matrix operand order repaired and checked by reading, not compiled, not run on hardware
//
#include <hip/hip_runtime.h>
#include <hip/hip_bf16.h>

typedef __attribute__((ext_vector_type(16))) _Float16 v16h;
typedef __attribute__((ext_vector_type(8)))  _Float16 v8h;
typedef __attribute__((ext_vector_type(8)))  float    v8f;

#define DD   128
#define HH   8
#define CC   16
#define NEG_SLOPE 0.2f
#define LN_EPS 1e-5f

// wfrag halves per matrix: 4 ktiles * 8 coltiles * 32 lanes * 16 elems
#define WFRAG_PER_MAT (4 * 8 * 32 * 16)

static __device__ __forceinline__ float leaky(float v) {
    return v > 0.f ? v : NEG_SLOPE * v;
}

static __device__ __forceinline__ float wave_sum(float v) {
#pragma unroll
    for (int m = 16; m > 0; m >>= 1) v += __shfl_xor(v, m, 32);
    return v;
}

// ---------------- f16 staging ----------------

__global__ void k_convert_x(const float* __restrict__ x, _Float16* __restrict__ xh, int total) {
    int i = blockIdx.x * 256 + threadIdx.x;
    if (i < total) xh[i] = (_Float16)x[i];
}

// Pack Ws [3,128,128] into WMMA-B fragment-native layout:
// wfrag[(((mat*4+kt)*8+ct)*32+lane)*16 + e] = W[mat][kt*32 + hi*16 + e][ct*16 + n]
// with n = lane&15, hi = lane>>4  (contiguous k order used here, same as A: lanes 0-15 hold K=0..15, 16-31 hold K=16..31)
__global__ void k_pack_w(const float* __restrict__ Ws, _Float16* __restrict__ wfrag) {
    int i = blockIdx.x * 256 + threadIdx.x;
    if (i >= 3 * WFRAG_PER_MAT) return;
    int e    = i & 15;
    int lane = (i >> 4) & 31;
    int ct   = (i >> 9) & 7;
    int kt   = (i >> 12) & 3;
    int mat  = i >> 14;
    int n  = lane & 15;
    int hi = lane >> 4;
    int k   = kt * 32 + hi * 16 + e;
    int col = ct * 16 + n;
    wfrag[i] = (_Float16)Ws[((size_t)mat * DD + k) * DD + col];
}

// ---------------- WMMA GEMM: h[mat] = x @ Ws[mat] (+bias for mat 0) ----------------
// block = 256 (8 waves); each wave computes a 16-row x 128-col strip.
// W fragments for this mat are staged once in LDS; A fragments hoisted into registers.
__global__ void k_gemm_wmma(const _Float16* __restrict__ xh,
                            const _Float16* __restrict__ wfrag,
                            const float* __restrict__ gbias,
                            float* __restrict__ h, int N) {
    __shared__ _Float16 lw[WFRAG_PER_MAT];   // 32 KB

    int mat  = blockIdx.y;
    int wave = threadIdx.x >> 5;
    int lane = threadIdx.x & 31;
    int m  = lane & 15;
    int hi = lane >> 4;
    int rowbase = blockIdx.x * 128 + wave * 16;
    int arow = rowbase + m;
    int arowc = arow < N ? arow : (N - 1);

    // cooperative W-frag stage: 32 KB / 256 threads = 8 x 16B per thread
    {
        const int4* src = (const int4*)(wfrag + (size_t)mat * WFRAG_PER_MAT);
        int4* dst = (int4*)lw;
#pragma unroll
        for (int j = 0; j < 8; ++j) {
            int idx = threadIdx.x + j * 256;
            dst[idx] = src[idx];
        }
    }

    // hoist all 4 A fragments (one 16x32 tile each): elems 0..7 = K=kbase+hi*16+0..7,
    // elems 8..15 = K=kbase+hi*16+8+0..7
    v16h afrag[4];
#pragma unroll
    for (int kt = 0; kt < 4; ++kt) {
        const _Float16* ap = xh + (size_t)arowc * DD + kt * 32 + hi * 16;
        v8h alo = *(const v8h*)(ap);
        v8h ahi = *(const v8h*)(ap + 8);
#pragma unroll
        for (int j = 0; j < 8; ++j) { afrag[kt][j] = alo[j]; afrag[kt][8 + j] = ahi[j]; }
    }

    __syncthreads();

    v8f acc[8] = {};
#pragma unroll
    for (int kt = 0; kt < 4; ++kt) {
#pragma unroll
        for (int ct = 0; ct < 8; ++ct) {
            v16h bfrag = *(const v16h*)(lw + (((kt * 8) + ct) * 32 + lane) * 16);
            acc[ct] = __builtin_amdgcn_wmma_f32_16x16x32_f16(
                false, afrag[kt], false, bfrag, (short)0, acc[ct], false, false);
        }
    }

    // C/D layout: VGPR r holds C[row = rowbase + r + 8*hi, col = ct*16 + (lane&15)]
#pragma unroll
    for (int ct = 0; ct < 8; ++ct) {
        int col = ct * 16 + m;
        float b = (mat == 0) ? gbias[col] : 0.f;
#pragma unroll
        for (int r = 0; r < 8; ++r) {
            int row = rowbase + r + 8 * hi;
            if (row < N)
                h[((size_t)mat * N + row) * DD + col] = acc[ct][r] + b;
        }
    }
}

// ---------------- attention coefficients + m init (self-loop logit) ----------------
__global__ void k_attn_coef(const float* __restrict__ h,
                            const float* __restrict__ asrc,
                            const float* __restrict__ adst,
                            float* __restrict__ es, float* __restrict__ ed,
                            float* __restrict__ mx, int N) {
    int i = blockIdx.x * 256 + threadIdx.x;
    if (i >= 2 * N * HH) return;
    int hd = i & 7;
    int n  = (i >> 3) % N;
    int g  = i / (N * HH);
    const float* hrow = h + ((size_t)(g + 1) * N + n) * DD + hd * CC;
    const float* as = asrc + (size_t)(g + 1) * DD + hd * CC;
    const float* ad = adst + (size_t)(g + 1) * DD + hd * CC;
    float s = 0.f, d = 0.f;
#pragma unroll
    for (int c = 0; c < CC; ++c) { s += hrow[c] * as[c]; d += hrow[c] * ad[c]; }
    size_t o = ((size_t)g * N + n) * HH + hd;
    es[o] = s;
    ed[o] = d;
    mx[o] = leaky(s + d);   // self-loop edge logit initializes segment max
}

// ---------------- segment max over incoming edges ----------------
__global__ void k_edge_max(const int* __restrict__ top, const int* __restrict__ bot,
                           const float* __restrict__ es, const float* __restrict__ ed,
                           float* __restrict__ mx, int N, int K) {
    int t = blockIdx.x * 256 + threadIdx.x;
    if (t >= 2 * N * K) return;
    int k = t % K;
    int n = (t / K) % N;
    int g = t / (N * K);
    const int* idxp = g ? bot : top;
    int dst = idxp[(size_t)n * K + k];
    if (dst == n) return;   // remove_self_loops: invalid edge
    size_t so = ((size_t)g * N + n) * HH;
    size_t dofs = ((size_t)g * N + dst) * HH;
#pragma unroll
    for (int hd = 0; hd < HH; ++hd) {
        float e = leaky(es[so + hd] + ed[dofs + hd]);
        atomicMax(&mx[dofs + hd], e);
    }
}

// ---------------- self-loop contribution initializes denom & numer ----------------
__global__ void k_self_init(const float* __restrict__ h,
                            const float* __restrict__ es, const float* __restrict__ ed,
                            const float* __restrict__ mx,
                            float* __restrict__ den, float* __restrict__ num, int N) {
    int i = blockIdx.x * 256 + threadIdx.x;
    if (i >= 2 * N * DD) return;
    int c = i & (DD - 1);
    int n = (i >> 7) % N;
    int g = i / (N * DD);
    int hd = c >> 4;
    size_t ho = ((size_t)g * N + n) * HH + hd;
    float e0 = leaky(es[ho] + ed[ho]);
    float ex = __expf(e0 - mx[ho]);
    num[((size_t)g * N + n) * DD + c] = ex * h[((size_t)(g + 1) * N + n) * DD + c];
    if ((c & 15) == 0) den[ho] = ex;
}

// ---------------- edge pass: accumulate exp weights & weighted features ----------------
// one lane handles 4 channels of one (g, edge)
__global__ void k_edge_acc(const int* __restrict__ top, const int* __restrict__ bot,
                           const float* __restrict__ h,
                           const float* __restrict__ es, const float* __restrict__ ed,
                           const float* __restrict__ mx,
                           float* __restrict__ den, float* __restrict__ num,
                           int N, int K) {
    int t = blockIdx.x * 256 + threadIdx.x;
    int NK = N * K;
    if (t >= 2 * NK * 32) return;
    int lane = t & 31;
    int e    = (t >> 5) % NK;
    int g    = (t >> 5) / NK;
    int k = e % K;
    int n = e / K;   // src
    const int* idxp = g ? bot : top;
    int dst = idxp[(size_t)n * K + k];
    if (dst == n) return;  // invalid edge
    int c0 = lane * 4;
    int hd = c0 >> 4;
    float logit = leaky(es[((size_t)g * N + n) * HH + hd] +
                        ed[((size_t)g * N + dst) * HH + hd]);
    float ex = __expf(logit - mx[((size_t)g * N + dst) * HH + hd]);
    if ((c0 & 15) == 0)
        atomicAdd(&den[((size_t)g * N + dst) * HH + hd], ex);
    const float* hr = h + ((size_t)(g + 1) * N + n) * DD + c0;
    float* np = num + ((size_t)g * N + dst) * DD + c0;
#pragma unroll
    for (int j = 0; j < 4; ++j) atomicAdd(np + j, ex * hr[j]);
}

// ---------------- per-node epilogue: gate softmax, GELU, residual, LayerNorm ----------------
// one wave32 per node; each lane owns 4 channels.
__global__ void k_finalize(const float* __restrict__ x,
                           const float* __restrict__ h,       // h[0] = self_out (bias folded)
                           const float* __restrict__ num, const float* __restrict__ den,
                           const float* __restrict__ gbias,
                           const float* __restrict__ impw, const float* __restrict__ impb,
                           const float* __restrict__ lng, const float* __restrict__ lnb,
                           float* __restrict__ out, int N) {
    int wave = threadIdx.x >> 5;
    int lane = threadIdx.x & 31;
    int n = blockIdx.x * 8 + wave;
    if (n >= N) return;
    int c0 = lane * 4;
    int hd = c0 >> 4;

    float sf[4], tp[4], bt[4], xr[4];
    float d_top = den[(size_t)n * HH + hd];
    float d_bot = den[((size_t)N + n) * HH + hd];
#pragma unroll
    for (int j = 0; j < 4; ++j) {
        int c = c0 + j;
        sf[j] = h[(size_t)n * DD + c];
        tp[j] = num[(size_t)n * DD + c] / d_top + gbias[DD + c];
        bt[j] = num[((size_t)N + n) * DD + c] / d_bot + gbias[2 * DD + c];
        xr[j] = x[(size_t)n * DD + c];
    }

    float p0 = 0.f, p1 = 0.f, p2 = 0.f;
#pragma unroll
    for (int j = 0; j < 4; ++j) {
        int c = c0 + j;
        p0 += sf[j] * impw[c];
        p1 += tp[j] * impw[DD + c];
        p2 += bt[j] * impw[2 * DD + c];
    }
    p0 = wave_sum(p0) + impb[0];
    p1 = wave_sum(p1) + impb[1];
    p2 = wave_sum(p2) + impb[2];

    float mxv = fmaxf(p0, fmaxf(p1, p2));
    float e0 = __expf(p0 - mxv), e1 = __expf(p1 - mxv), e2 = __expf(p2 - mxv);
    float inv = 1.f / (e0 + e1 + e2);
    float w0 = e0 * inv, w1 = e1 * inv, w2 = e2 * inv;

    if (lane == 0) {
        float* wout = out + (size_t)N * DD + (size_t)n * 3;
        wout[0] = w0; wout[1] = w1; wout[2] = w2;
    }

    float y[4];
    float psum = 0.f;
#pragma unroll
    for (int j = 0; j < 4; ++j) {
        float v = w0 * sf[j] + w1 * tp[j] + w2 * bt[j];
        float gelu = 0.5f * v * (1.0f + erff(v * 0.70710678118654752f));
        y[j] = gelu + xr[j];
        psum += y[j];
    }
    float mu = wave_sum(psum) * (1.0f / DD);
    float vsum = 0.f;
#pragma unroll
    for (int j = 0; j < 4; ++j) { float d = y[j] - mu; vsum += d * d; }
    float var = wave_sum(vsum) * (1.0f / DD);
    float rstd = rsqrtf(var + LN_EPS);
#pragma unroll
    for (int j = 0; j < 4; ++j) {
        int c = c0 + j;
        out[(size_t)n * DD + c] = (y[j] - mu) * rstd * lng[c] + lnb[c];
    }
}

static inline size_t align256(size_t v) { return (v + 255) & ~(size_t)255; }

extern "C" void kernel_launch(void* const* d_in, const int* in_sizes, int n_in,
                              void* d_out, int out_size, void* d_ws, size_t ws_size,
                              hipStream_t stream) {
    const float* x    = (const float*)d_in[0];
    const int*   top  = (const int*)d_in[1];
    const int*   bot  = (const int*)d_in[2];
    const float* Ws   = (const float*)d_in[3];
    const float* asrc = (const float*)d_in[4];
    const float* adst = (const float*)d_in[5];
    const float* gbias= (const float*)d_in[6];
    const float* impw = (const float*)d_in[7];
    const float* impb = (const float*)d_in[8];
    const float* lng  = (const float*)d_in[9];
    const float* lnb  = (const float*)d_in[10];
    float* out = (float*)d_out;

    const int N = in_sizes[0] / DD;
    const int K = in_sizes[1] / N;

    char* p = (char*)d_ws;
    _Float16* xh = (_Float16*)p;  p += align256((size_t)N * DD * sizeof(_Float16));
    _Float16* wfrag = (_Float16*)p; p += align256((size_t)3 * WFRAG_PER_MAT * sizeof(_Float16));
    float* h   = (float*)p; p += align256((size_t)3 * N * DD * sizeof(float));
    float* es  = (float*)p; p += align256((size_t)2 * N * HH * sizeof(float));
    float* ed  = (float*)p; p += align256((size_t)2 * N * HH * sizeof(float));
    float* mx  = (float*)p; p += align256((size_t)2 * N * HH * sizeof(float));
    float* den = (float*)p; p += align256((size_t)2 * N * HH * sizeof(float));
    float* num = (float*)p; p += align256((size_t)2 * N * DD * sizeof(float));

    {
        int tot = N * DD;
        k_convert_x<<<(tot + 255) / 256, 256, 0, stream>>>(x, xh, tot);
    }
    k_pack_w<<<(3 * WFRAG_PER_MAT + 255) / 256, 256, 0, stream>>>(Ws, wfrag);

    {
        dim3 grid((N + 127) / 128, 3);
        k_gemm_wmma<<<grid, 256, 0, stream>>>(xh, wfrag, gbias, h, N);
    }
    {
        int tot = 2 * N * HH;
        k_attn_coef<<<(tot + 255) / 256, 256, 0, stream>>>(h, asrc, adst, es, ed, mx, N);
    }
    {
        int tot = 2 * N * K;
        k_edge_max<<<(tot + 255) / 256, 256, 0, stream>>>(top, bot, es, ed, mx, N, K);
    }
    {
        int tot = 2 * N * DD;
        k_self_init<<<(tot + 255) / 256, 256, 0, stream>>>(h, es, ed, mx, den, num, N);
    }
    {
        long long tot = (long long)2 * N * K * 32;
        k_edge_acc<<<(int)((tot + 255) / 256), 256, 0, stream>>>(top, bot, h, es, ed, mx,
                                                                 den, num, N, K);
    }
    {
        int grid = (N + 7) / 8;
        k_finalize<<<grid, 256, 0, stream>>>(x, h, num, den, gbias, impw, impb,
                                             lng, lnb, out, N);
    }
}
